// TransConv_60894046322855
// MI455X (gfx1250) — matrix operand order repaired and checked by reading, not compiled, not run on hardware
//
#include <hip/hip_runtime.h>
#include <hip/hip_bf16.h>

typedef __bf16 bf16;
typedef bf16  v16bf __attribute__((ext_vector_type(16)));
typedef float v8f   __attribute__((ext_vector_type(8)));
typedef int   v4i_t __attribute__((ext_vector_type(4)));

#define KDIM 256   // in_channels
#define HEADS 8
#define DH 64      // per-head dim
#define HD 512     // HEADS*DH

union FragBF { v16bf v; uint4 u[2]; };

// ---- CDNA5 async global->LDS copy (ASYNCcnt path), with safe fallback -----
#if defined(__HIP_DEVICE_COMPILE__) && __has_builtin(__builtin_amdgcn_global_load_async_to_lds_b128)
#define USE_ASYNC_LDS 1
#define AS1V4(p) ((__attribute__((address_space(1))) v4i_t*)(p))
#define AS3V4(p) ((__attribute__((address_space(3))) v4i_t*)(p))
#define ASYNC_CP_B128(g, l) \
    __builtin_amdgcn_global_load_async_to_lds_b128(AS1V4(g), AS3V4(l), 0, 0)
#if __has_builtin(__builtin_amdgcn_s_wait_asynccnt)
#define ASYNC_WAIT(n) __builtin_amdgcn_s_wait_asynccnt(n)
#else
#define ASYNC_WAIT(n) asm volatile("s_wait_asynccnt %0" :: "i"(n) : "memory")
#endif
#endif

// ---------------------------------------------------------------- utilities
__global__ __launch_bounds__(256) void zero_f32_k(float* __restrict__ p, long n) {
    long i = (long)blockIdx.x * 256 + threadIdx.x;
    if (i < n) p[i] = 0.0f;
}

__global__ __launch_bounds__(256) void cvt_f32_bf16_k(const float* __restrict__ in,
                                                      bf16* __restrict__ out, long n) {
    long i = (long)blockIdx.x * 256 + threadIdx.x;
    if (i < n) out[i] = (bf16)in[i];
}

// Wt[n*K + k] = (bf16)W[k*Nout + n]
__global__ __launch_bounds__(256) void transpose_cvt_k(const float* __restrict__ W,
                                                       bf16* __restrict__ Wt,
                                                       int K, int Nout) {
    int i = blockIdx.x * 256 + threadIdx.x;
    if (i < K * Nout) {
        int k = i / Nout;
        int n = i - k * Nout;
        Wt[(size_t)n * K + k] = (bf16)W[i];
    }
}

__global__ __launch_bounds__(256) void degree_k(const int* __restrict__ col,
                                                float* __restrict__ deg, int E) {
    int i = blockIdx.x * 256 + threadIdx.x;
    if (i < E) atomicAdd(&deg[col[i]], 1.0f);
}

// ---------------------------------------------------------------- WMMA GEMM
// C[M,Nout] = Xbf[M,256] * Wt[Nout,256]^T + bias.  Block tile 64Mx64N,
// 8 waves: wave = (msub 0..3) x (nsub 0..1), each wave does 2 wmma per K step.
// Staging uses double-buffered GLOBAL_LOAD_ASYNC_TO_LDS_B128 when available.
__global__ __launch_bounds__(256)
void wmma_gemm_k(const bf16* __restrict__ X, const bf16* __restrict__ Wt,
                 const float* __restrict__ bias, void* __restrict__ Cout,
                 int M, int Nout, int out_is_bf16) {
    __shared__ bf16 sA[2][64 * 32];
    __shared__ bf16 sB[2][64 * 32];
    const int tid  = threadIdx.x;
    const int lane = tid & 31;
    const int half = (lane >> 4) & 1;
    const int r    = lane & 15;
    const int wave = tid >> 5;
    const int msub = wave >> 1;   // 0..3
    const int nsub = wave & 1;    // 0..1
    const int m0 = blockIdx.y * 64;
    const int n0 = blockIdx.x * 64;

    v8f acc0 = {}; v8f acc1 = {};

    // staging: each thread moves one uint4 of A and one of B per K-step
    const int ldRow   = tid >> 2;          // 0..63
    const int ldChunk = (tid & 3) << 3;    // bf16 elements: 0,8,16,24
    const int slot    = ldRow * 32 + ldChunk;
    int aRow = m0 + ldRow; if (aRow >= M) aRow = M - 1;   // clamp (stores guarded)
    const bf16* aPtr = X  + (size_t)aRow * KDIM + ldChunk;
    const bf16* bPtr = Wt + (size_t)(n0 + ldRow) * KDIM + ldChunk;

#ifdef USE_ASYNC_LDS
    // prefetch tile 0 into buffer 0
    ASYNC_CP_B128(aPtr, &sA[0][slot]);
    ASYNC_CP_B128(bPtr, &sB[0][slot]);
#endif

    for (int k0 = 0; k0 < KDIM; k0 += 32) {
        const int buf = (k0 >> 5) & 1;
#ifdef USE_ASYNC_LDS
        if (k0 + 32 < KDIM) {
            // prefetch next tile into the other buffer (free since iter k-1's
            // trailing barrier), then wait for current tile (loads retire in
            // order, so asynccnt<=2 leaves only the 2 prefetch ops in flight)
            ASYNC_CP_B128(aPtr + k0 + 32, &sA[buf ^ 1][slot]);
            ASYNC_CP_B128(bPtr + k0 + 32, &sB[buf ^ 1][slot]);
            ASYNC_WAIT(2);
        } else {
            ASYNC_WAIT(0);
        }
#else
        *(uint4*)&sA[buf][slot] = *(const uint4*)(aPtr + k0);
        *(uint4*)&sB[buf][slot] = *(const uint4*)(bPtr + k0);
#endif
        __syncthreads();

        // A fragment: lane(half,r): contiguous K run half*16..half*16+15
        FragBF fa;
        {
            const bf16* ab = &sA[buf][(msub * 16) * 32];
            fa.u[0] = *(const uint4*)(ab + r * 32 + half * 16);
            fa.u[1] = *(const uint4*)(ab + r * 32 + half * 16 + 8);
        }
        // B fragments: lane(half,r): contiguous K run half*16..half*16+15
        FragBF fb0, fb1;
        {
            const bf16* bb0 = &sB[buf][(nsub * 32) * 32];
            fb0.u[0] = *(const uint4*)(bb0 + r * 32 + half * 16);
            fb0.u[1] = *(const uint4*)(bb0 + r * 32 + half * 16 + 8);
            const bf16* bb1 = &sB[buf][(nsub * 32 + 16) * 32];
            fb1.u[0] = *(const uint4*)(bb1 + r * 32 + half * 16);
            fb1.u[1] = *(const uint4*)(bb1 + r * 32 + half * 16 + 8);
        }
        acc0 = __builtin_amdgcn_wmma_f32_16x16x32_bf16(false, fa.v, false, fb0.v,
                                                       (short)0, acc0, false, false);
        acc1 = __builtin_amdgcn_wmma_f32_16x16x32_bf16(false, fa.v, false, fb1.v,
                                                       (short)0, acc1, false, false);
        __syncthreads();
    }

    // C layout: VGPR j -> M = j + half*8 ; N = r
    const int mBase = m0 + msub * 16 + half * 8;
    const int nA = n0 + nsub * 32 + r;
    const int nB = nA + 16;
    const float biasA = bias[nA];
    const float biasB = bias[nB];
#pragma unroll
    for (int j = 0; j < 8; ++j) {
        int m = mBase + j;
        if (m < M) {
            float va = acc0[j] + biasA;
            float vb = acc1[j] + biasB;
            if (out_is_bf16) {
                bf16* C = (bf16*)Cout;
                C[(size_t)m * Nout + nA] = (bf16)va;
                C[(size_t)m * Nout + nB] = (bf16)vb;
            } else {
                float* C = (float*)Cout;
                C[(size_t)m * Nout + nA] = va;
                C[(size_t)m * Nout + nB] = vb;
            }
        }
    }
}

// ---------------------------------------------------------------- kvs reduce
// kvs[h][m][d] = sum_n (k[n,h,m]/||k||) * v[n,h,d]; ksum[h][m]; vsum[h][d]
// grid = (8 heads, node chunks of 1024), block = 256 threads.
__global__ __launch_bounds__(256)
void kvs_reduce_k(const bf16* __restrict__ kbf, const bf16* __restrict__ vbf,
                  float* __restrict__ kvs, float* __restrict__ ksum,
                  float* __restrict__ vsum, int Nn) {
    __shared__ bf16  skt[64 * 64];
    __shared__ bf16  svt[64 * 64];
    __shared__ float rnorm[64];
    const int tid = threadIdx.x;
    const int h   = blockIdx.x;
    const int d   = tid & 63;
    const int mg  = tid >> 6;      // 0..3 (owns m = mg*16 + j)
    float acc[16];
#pragma unroll
    for (int j = 0; j < 16; ++j) acc[j] = 0.0f;
    float ksacc = 0.0f, vsacc = 0.0f;

    const int nodeBase = blockIdx.y * 1024;
    for (int t = 0; t < 16; ++t) {
        const int nb = nodeBase + t * 64;
#pragma unroll
        for (int s = 0; s < 2; ++s) {
            int idx  = tid + s * 256;        // 0..511
            int node = idx >> 3;
            int ch   = (idx & 7) << 3;       // bf16 elems
            int gn   = nb + node;
            uint4 kq, vq;
            if (gn < Nn) {
                kq = *(const uint4*)(kbf + (size_t)gn * HD + h * 64 + ch);
                vq = *(const uint4*)(vbf + (size_t)gn * HD + h * 64 + ch);
            } else {
                kq.x = kq.y = kq.z = kq.w = 0u;
                vq = kq;
            }
            *(uint4*)&skt[node * 64 + ch] = kq;
            *(uint4*)&svt[node * 64 + ch] = vq;
        }
        __syncthreads();
        if (tid < 64) {
            float s = 0.0f;
            for (int i = 0; i < 64; ++i) { float x = (float)skt[tid * 64 + i]; s += x * x; }
            rnorm[tid] = (s > 0.0f) ? rsqrtf(s) : 0.0f;
        }
        __syncthreads();
        for (int node = 0; node < 64; ++node) {
            float rv = rnorm[node];
            float vv = (float)svt[node * 64 + d];
            float vr = vv * rv;
#pragma unroll
            for (int j = 0; j < 16; ++j)
                acc[j] += (float)skt[node * 64 + mg * 16 + j] * vr;
            if (mg == 0) {
                ksacc += (float)skt[node * 64 + d] * rv;
                vsacc += vv;
            }
        }
        __syncthreads();
    }
#pragma unroll
    for (int j = 0; j < 16; ++j)
        atomicAdd(&kvs[h * 4096 + (mg * 16 + j) * 64 + d], acc[j]);
    if (mg == 0) {
        atomicAdd(&ksum[h * 64 + d], ksacc);
        atomicAdd(&vsum[h * 64 + d], vsacc);
    }
}

// ------------------------------------------------------- attention epilogue
// out[n,d] = mean_h( (q̂·kvs_h + vsum_h)[d] / (q̂·ksum_h + N) ) + residual[n,d]
__global__ __launch_bounds__(256)
void attn_epilogue_k(const bf16* __restrict__ qbf, const float* __restrict__ kvs,
                     const float* __restrict__ ksum, const float* __restrict__ vsum,
                     const float* __restrict__ rres, float* __restrict__ out,
                     int Nn, int npb) {
    extern __shared__ float sm[];
    float* skvs  = sm;                   // 32768
    float* sks   = skvs + 32768;         // 512
    float* svs   = sks + 512;            // 512
    float* sq    = svs + 512;            // 512
    float* rqn   = sq + 512;             // 8
    float* sden  = rqn + 8;              // 8
    float* hpart = sden + 8;             // 256
    const int tid = threadIdx.x;

    for (int i = tid; i < 32768; i += 256) skvs[i] = kvs[i];
    for (int i = tid; i < 512; i += 256) { sks[i] = ksum[i]; svs[i] = vsum[i]; }
    __syncthreads();

    const int d  = tid & 63;
    const int mg = tid >> 6;
    int n0 = blockIdx.x * npb;
    int n1 = n0 + npb; if (n1 > Nn) n1 = Nn;
    for (int n = n0; n < n1; ++n) {
        for (int i = tid; i < HD; i += 256) sq[i] = (float)qbf[(size_t)n * HD + i];
        __syncthreads();
        if (tid < 8) {
            float s = 0.0f, dn = 0.0f;
            for (int i = 0; i < 64; ++i) {
                float x = sq[tid * 64 + i];
                s  += x * x;
                dn += x * sks[tid * 64 + i];
            }
            float rq = (s > 0.0f) ? rsqrtf(s) : 0.0f;
            rqn[tid]  = rq;
            sden[tid] = rq * dn + (float)Nn;
        }
        __syncthreads();
        float a = 0.0f;
#pragma unroll
        for (int hh = 0; hh < 2; ++hh) {
            int h = mg + hh * 4;
            const float* kb = &skvs[h * 4096 + d];
            const float* qb = &sq[h * 64];
            float num = 0.0f;
            for (int m = 0; m < 64; ++m) num += qb[m] * kb[m * 64];
            num = num * rqn[h] + svs[h * 64 + d];
            a += num / sden[h];
        }
        hpart[tid] = a;
        __syncthreads();
        if (tid < 64) {
            float tot = hpart[tid] + hpart[tid + 64] + hpart[tid + 128] + hpart[tid + 192];
            out[(size_t)n * 64 + tid] = tot * 0.125f + rres[(size_t)n * 64 + tid];
        }
        __syncthreads();
    }
}

// ---------------------------------------------------------------- GNN edges
__global__ __launch_bounds__(256)
void gnn_edge_k(const int* __restrict__ ei, const float* __restrict__ ew,
                const float* __restrict__ deg, const bf16* __restrict__ gbf,
                float* __restrict__ out, int E) {
    long gid = (long)blockIdx.x * 256 + threadIdx.x;
    int e = (int)(gid >> 6);
    int d = (int)(gid & 63);
    if (e >= E) return;
    int row = ei[e];
    int col = ei[(size_t)E + e];
    float val = ew[e] * rsqrtf(deg[col]) * rsqrtf(deg[row]);
    if (!__builtin_isfinite(val)) val = 0.0f;   // nan_to_num
    atomicAdd(&out[(size_t)col * 64 + d], val * (float)gbf[(size_t)row * 64 + d]);
}

// ---------------------------------------------------------------- launcher
extern "C" void kernel_launch(void* const* d_in, const int* in_sizes, int n_in,
                              void* d_out, int out_size, void* d_ws, size_t ws_size,
                              hipStream_t stream) {
    const float* x_q = (const float*)d_in[0];
    const float* x_s = (const float*)d_in[1];
    const int*   ei  = (const int*)d_in[2];
    const float* ew  = (const float*)d_in[3];
    const float* Wq  = (const float*)d_in[4];  const float* bq = (const float*)d_in[5];
    const float* Wk  = (const float*)d_in[6];  const float* bk = (const float*)d_in[7];
    const float* Wv  = (const float*)d_in[8];  const float* bv = (const float*)d_in[9];
    const float* Wg  = (const float*)d_in[10]; const float* bg = (const float*)d_in[11];
    const float* Wr  = (const float*)d_in[12]; const float* br = (const float*)d_in[13];
    const int Nn = in_sizes[0] / KDIM;
    const int E  = in_sizes[3];
    float* out = (float*)d_out;

    char* ws = (char*)d_ws;
    size_t off = 0;
    auto alloc = [&](size_t bytes) -> char* {
        char* p = ws + off;
        off = (off + bytes + 255) & ~(size_t)255;
        return p;
    };
    size_t zBeg = off;
    float* deg  = (float*)alloc((size_t)Nn * 4);
    float* kvs  = (float*)alloc((size_t)HEADS * DH * DH * 4);
    float* ksum = (float*)alloc((size_t)HEADS * DH * 4);
    float* vsum = (float*)alloc((size_t)HEADS * DH * 4);
    size_t zEnd = off;
    bf16* Xq  = (bf16*)alloc((size_t)Nn * KDIM * 2);
    bf16* Xs  = (bf16*)alloc((size_t)Nn * KDIM * 2);
    bf16* Wqt = (bf16*)alloc((size_t)HD * KDIM * 2);
    bf16* Wkt = (bf16*)alloc((size_t)HD * KDIM * 2);
    bf16* Wvt = (bf16*)alloc((size_t)HD * KDIM * 2);
    bf16* Wgt = (bf16*)alloc((size_t)DH * KDIM * 2);
    bf16* Wrt = (bf16*)alloc((size_t)DH * KDIM * 2);
    bf16* qb  = (bf16*)alloc((size_t)Nn * HD * 2);
    bf16* kb  = (bf16*)alloc((size_t)Nn * HD * 2);
    bf16* vb  = (bf16*)alloc((size_t)Nn * HD * 2);
    bf16* gb  = (bf16*)alloc((size_t)Nn * DH * 2);
    float* rr = (float*)alloc((size_t)Nn * DH * 4);

    // 1) zero deg/kvs/ksum/vsum
    long zN = (long)((zEnd - zBeg) / 4);
    zero_f32_k<<<(int)((zN + 255) / 256), 256, 0, stream>>>((float*)(ws + zBeg), zN);

    // 2) in-degree
    degree_k<<<(E + 255) / 256, 256, 0, stream>>>(ei + (size_t)E, deg, E);

    // 3) f32 -> bf16 activations
    long nIn = (long)Nn * KDIM;
    cvt_f32_bf16_k<<<(int)((nIn + 255) / 256), 256, 0, stream>>>(x_q, Xq, nIn);
    cvt_f32_bf16_k<<<(int)((nIn + 255) / 256), 256, 0, stream>>>(x_s, Xs, nIn);

    // 4) weight transpose + convert
    transpose_cvt_k<<<(KDIM * HD + 255) / 256, 256, 0, stream>>>(Wq, Wqt, KDIM, HD);
    transpose_cvt_k<<<(KDIM * HD + 255) / 256, 256, 0, stream>>>(Wk, Wkt, KDIM, HD);
    transpose_cvt_k<<<(KDIM * HD + 255) / 256, 256, 0, stream>>>(Wv, Wvt, KDIM, HD);
    transpose_cvt_k<<<(KDIM * DH + 255) / 256, 256, 0, stream>>>(Wg, Wgt, KDIM, DH);
    transpose_cvt_k<<<(KDIM * DH + 255) / 256, 256, 0, stream>>>(Wr, Wrt, KDIM, DH);

    // 5) WMMA GEMMs
    dim3 gBig(HD / 64, (Nn + 63) / 64);
    dim3 gSm(DH / 64, (Nn + 63) / 64);
    wmma_gemm_k<<<gBig, 256, 0, stream>>>(Xq, Wqt, bq, (void*)qb, Nn, HD, 1);
    wmma_gemm_k<<<gBig, 256, 0, stream>>>(Xs, Wkt, bk, (void*)kb, Nn, HD, 1);
    wmma_gemm_k<<<gBig, 256, 0, stream>>>(Xs, Wvt, bv, (void*)vb, Nn, HD, 1);
    wmma_gemm_k<<<gSm, 256, 0, stream>>>(Xs, Wgt, bg, (void*)gb, Nn, DH, 1);
    wmma_gemm_k<<<gSm, 256, 0, stream>>>(Xs, Wrt, br, (void*)rr, Nn, DH, 0);

    // 6) kvs / ksum / vsum reduction
    dim3 gKvs(HEADS, (Nn + 1023) / 1024);
    kvs_reduce_k<<<gKvs, 256, 0, stream>>>(kb, vb, kvs, ksum, vsum, Nn);

    // 7) attention epilogue + residual (128KB kvs in LDS)
    const int npb = 256;
    size_t shBytes = (size_t)(32768 + 512 + 512 + 512 + 8 + 8 + 256) * sizeof(float);
    attn_epilogue_k<<<(Nn + npb - 1) / npb, 256, shBytes, stream>>>(
        qb, kvs, ksum, vsum, rr, out, Nn, npb);

    // 8) sparse GCN scatter (L2-resident atomics)
    long eg = (long)E * 64;
    gnn_edge_k<<<(int)((eg + 255) / 256), 256, 0, stream>>>(ei, ew, deg, gb, out, E);
}
